// LineAGC_66503273611803
// MI455X (gfx1250) — matrix operand order repaired and checked by reading, not compiled, not run on hardware
//
#include <hip/hip_runtime.h>

// Problem constants (from reference): B=32, T=24 -> BT=768; M=512; Din=Dout=64.
#define BT   768
#define MN   512
#define DIN  64
#define DOUT 64

typedef __attribute__((ext_vector_type(8)))  _Float16 v8h;
typedef __attribute__((ext_vector_type(16))) _Float16 v16h;
typedef __attribute__((ext_vector_type(8)))  float    v8f;

static __device__ __forceinline__ unsigned short f2h(float f) {
    _Float16 h = (_Float16)f;             // single v_cvt_f16_f32
    return __builtin_bit_cast(unsigned short, h);
}
static __device__ __forceinline__ v8h ld8(const unsigned short* p) {
    return *(const v8h*)p;                // 16B aligned by construction
}
static __device__ __forceinline__ v16h cat8(v8h lo, v8h hi) {
    return __builtin_shufflevector(lo, hi, 0,1,2,3,4,5,6,7,8,9,10,11,12,13,14,15);
}

// ---------------------------------------------------------------------------
// Pass 0: adj fp32 -> fp16 ; route_weights [m][d][l] fp32 -> W_T[m][l][d] fp16
// ---------------------------------------------------------------------------
__global__ __launch_bounds__(256) void cvt_kernel(const float* __restrict__ adj,
                                                  const float* __restrict__ rw,
                                                  unsigned short* __restrict__ adjh,
                                                  unsigned short* __restrict__ wt) {
    const int stride = gridDim.x * blockDim.x;
    const int tid    = blockIdx.x * blockDim.x + threadIdx.x;
    for (int e = tid; e < MN * MN; e += stride) adjh[e] = f2h(adj[e]);
    for (int e = tid; e < MN * DIN * DOUT; e += stride) {
        int m = e >> 12, r = e & 4095, d = r >> 6, l = r & 63;
        wt[(m << 12) + (l << 6) + d] = f2h(rw[e]);
    }
}

// ---------------------------------------------------------------------------
// Pass 1: per-bt GEMM, computed as transposed tiles:
//   D'[d][m] = sum_n xT[d][n] * adj[m][n]   (A' = xT rows, B' = adj rows)
// so each lane's 8 C-elements are 8 consecutive d -> one b128 store into H.
// One workgroup (8 waves) per bt; x[bt] transposed into 64KB swizzled LDS.
// ---------------------------------------------------------------------------
__global__ __launch_bounds__(256) void gemm1_kernel(const unsigned short* __restrict__ adjh,
                                                    const float* __restrict__ x,
                                                    unsigned short* __restrict__ H) {
    // xT logical layout [d][n], stored in 16-element chunks with XOR swizzle:
    // idx(d,n) = (d<<9) + (((n>>4) ^ (d&31)) << 4) + (n&15)   -> 64KB total
    __shared__ unsigned short xT[DIN * MN];

    const int bt = blockIdx.x;
    const int t  = threadIdx.x;

    const float4* xb = (const float4*)(x + (size_t)bt * (MN * DIN));
    #pragma unroll 4
    for (int i = 0; i < 32; ++i) {
        int e4 = i * 256 + t;          // 8192 float4 loads, coalesced
        int n  = e4 >> 4;
        int d0 = (e4 & 15) << 2;
        float4 v = xb[e4];
        float vv[4] = {v.x, v.y, v.z, v.w};
        #pragma unroll
        for (int j = 0; j < 4; ++j) {
            int d = d0 + j;
            xT[(d << 9) + ((((n >> 4)) ^ (d & 31)) << 4) + (n & 15)] = f2h(vv[j]);
        }
    }
    __syncthreads();

    const int wave = t >> 5;
    const int lane = t & 31;
    const int half = lane >> 4;   // wave32 lane-group split per ISA fragment layout
    const int l16  = lane & 15;

    unsigned short* hbt = H + (size_t)bt * (MN * DIN);

    #pragma unroll 1
    for (int mt = wave; mt < MN / 16; mt += 8) {      // 4 m-tiles per wave
        const int m_base = mt << 4;
        v8f acc[4] = {};                               // 4 d-tiles kept live
        // B' fragment source: col N = m (= l16), K runs [k+8h..], [k+16+8h..] (n)
        const unsigned short* arow = adjh + (size_t)(m_base + l16) * MN + half * 8;
        #pragma unroll 2
        for (int k = 0; k < MN; k += 32) {
            v16h badj = cat8(ld8(arow + k), ld8(arow + k + 16));
            #pragma unroll
            for (int dt = 0; dt < 4; ++dt) {
                // A' fragment: row d, per-lane K runs [k+8h..] and [k+16+8h..]
                const int d    = (dt << 4) + l16;
                const int c0   = (k >> 4);
                const int idx0 = (d << 9) + (((c0)     ^ (d & 31)) << 4) + half * 8;
                const int idx1 = (d << 9) + (((c0 + 1) ^ (d & 31)) << 4) + half * 8;
                v16h ax = cat8(ld8(&xT[idx0]), ld8(&xT[idx1]));
                acc[dt] = __builtin_amdgcn_wmma_f32_16x16x32_f16(
                    false, ax, false, badj, (short)0, acc[dt], false, false);
            }
        }
        // D' rows = d (8 consecutive per lane), cols = m: packed 16B store
        #pragma unroll
        for (int dt = 0; dt < 4; ++dt) {
            v8h hv;
            #pragma unroll
            for (int j = 0; j < 8; ++j) hv[j] = (_Float16)acc[dt][j];
            unsigned short* p = hbt + (m_base + l16) * DIN + (dt << 4) + half * 8;
            *(v8h*)p = hv;
        }
    }
}

// ---------------------------------------------------------------------------
// Pass 2, transposed tiles:  D''[l][bt] = W_T[m](l x d) @ H^T (d x bt)
// Each lane's 8 C-elements are 8 consecutive l -> one 32B fp32 vector store;
// bias[m][l] becomes a per-lane 32B vector added elementwise.
// One workgroup (8 waves) per m; W fragments hoisted into registers.
// ---------------------------------------------------------------------------
__global__ __launch_bounds__(256) void gemm2_kernel(const unsigned short* __restrict__ H,
                                                    const unsigned short* __restrict__ WT,
                                                    const float* __restrict__ bias,
                                                    float* __restrict__ out) {
    const int m    = blockIdx.x;
    const int t    = threadIdx.x;
    const int wave = t >> 5;
    const int lane = t & 31;
    const int half = lane >> 4;
    const int l16  = lane & 15;

    // A'' fragments: W_T[m][l][d], row l = lt*16+l16, 16 contiguous K [32kk+16h..]
    const unsigned short* wtm = WT + ((size_t)m << 12);
    v16h wfrag[4][2];
    v8f  bvv[4];
    #pragma unroll
    for (int lt = 0; lt < 4; ++lt) {
        const unsigned short* col = wtm + ((lt << 4) + l16) * DIN + half * 16;
        #pragma unroll
        for (int kk = 0; kk < 2; ++kk)
            wfrag[lt][kk] = cat8(ld8(col + kk * 32), ld8(col + kk * 32 + 8));
        // 8 consecutive bias values for this lane's 8 output rows (l)
        bvv[lt] = *(const v8f*)(bias + m * DOUT + (lt << 4) + half * 8);
    }

    #pragma unroll 2
    for (int i = 0; i < 6; ++i) {                     // 6 bt-tiles per wave (8*6=48)
        const int bt_base = (wave * 6 + i) << 4;
        // B'' fragment: col bt = l16, 16 contiguous K values (d)
        const unsigned short* hrow =
            H + (size_t)(bt_base + l16) * (MN * DIN) + m * DIN + half * 16;
        v16h b0 = cat8(ld8(hrow),      ld8(hrow + 8));    // d = 0..31
        v16h b1 = cat8(ld8(hrow + 32), ld8(hrow + 40));   // d = 32..63
        v8f acc[4] = {};
        #pragma unroll
        for (int lt = 0; lt < 4; ++lt) {
            acc[lt] = __builtin_amdgcn_wmma_f32_16x16x32_f16(
                false, wfrag[lt][0], false, b0, (short)0, acc[lt], false, false);
            acc[lt] = __builtin_amdgcn_wmma_f32_16x16x32_f16(
                false, wfrag[lt][1], false, b1, (short)0, acc[lt], false, false);
        }
        #pragma unroll
        for (int lt = 0; lt < 4; ++lt) {
            v8f res = acc[lt] + bvv[lt];
            float* p = out + ((size_t)(bt_base + l16) * MN + m) * DOUT
                           + (lt << 4) + half * 8;
            *(v8f*)p = res;                            // 32B contiguous per lane
        }
    }
}

// ---------------------------------------------------------------------------
extern "C" void kernel_launch(void* const* d_in, const int* in_sizes, int n_in,
                              void* d_out, int out_size, void* d_ws, size_t ws_size,
                              hipStream_t stream) {
    (void)in_sizes; (void)n_in; (void)out_size; (void)ws_size;
    const float* x   = (const float*)d_in[0];   // [768,512,64]
    const float* adj = (const float*)d_in[1];   // [512,512]
    const float* rw  = (const float*)d_in[2];   // [512,64,64]
    const float* rb  = (const float*)d_in[3];   // [512,1,64]
    float* out = (float*)d_out;

    // Workspace layout: adj f16 (512KB) | W_T f16 (4MB) | H f16 (48MB)
    unsigned short* adjh = (unsigned short*)d_ws;
    unsigned short* WT   = adjh + (size_t)MN * MN;
    unsigned short* Hh   = WT + (size_t)MN * DIN * DOUT;

    cvt_kernel  <<<1024, 256, 0, stream>>>(adj, rw, adjh, WT);
    gemm1_kernel<<<BT,   256, 0, stream>>>(adjh, x, Hh);
    gemm2_kernel<<<MN,   256, 0, stream>>>(Hh, WT, rb, out);
}
